// GQA_57380763075168
// MI455X (gfx1250) — matrix operand order repaired and checked by reading, not compiled, not run on hardware
//
#include <hip/hip_runtime.h>
#include <hip/hip_bf16.h>

typedef __attribute__((ext_vector_type(16))) _Float16 v16h;
typedef __attribute__((ext_vector_type(8)))  _Float16 v8h;
typedef __attribute__((ext_vector_type(8)))  float    v8f;

// ---------------------------------------------------------------------------
// WMMA helpers (CDNA5 gfx1250, wave32)
// ---------------------------------------------------------------------------
__device__ __forceinline__ v8f wmma_f16(v16h a, v16h b, v8f c) {
  // (neg_a, A, neg_b, B, c_mod, C, reuse_a, reuse_b)
  return __builtin_amdgcn_wmma_f32_16x16x32_f16(false, a, false, b, (short)0, c,
                                                false, false);
}

// A-matrix fragment, 16x32 f16 (ISA 7.12.2): lane r (r=lane&15) holds row r.
// VGPR0..3 = K (hi*8)+0..7 pairs, VGPR4..7 = K 16+(hi*8)+0..7, hi = lane>>4.
__device__ __forceinline__ v16h frag_a(const _Float16* base, int stride, int lane) {
  const int r = lane & 15, hi = (lane >> 4) & 1;
  const _Float16* p = base + (size_t)r * stride + hi * 8;
  v8h lo = *(const v8h*)(p);
  v8h hh = *(const v8h*)(p + 16);
  return __builtin_shufflevector(lo, hh, 0, 1, 2, 3, 4, 5, 6, 7,
                                 8, 9, 10, 11, 12, 13, 14, 15);
}

// B-matrix fragment, 32x16 f16: lane n (n=lane&15) holds column n; elements
// 0..7 = K (hi*8)+0..7, elements 8..15 = K 16+(hi*8)+0..7 (same as A). Source is
// column-major storage Bt[n][k] (k contiguous), so this is a row load.
__device__ __forceinline__ v16h frag_b(const _Float16* base, int stride, int lane) {
  const int r = lane & 15, hi = (lane >> 4) & 1;
  const _Float16* p = base + (size_t)r * stride + hi * 8;
  v8h lo = *(const v8h*)(p);
  v8h hh = *(const v8h*)(p + 16);
  return __builtin_shufflevector(lo, hh, 0, 1, 2, 3, 4, 5, 6, 7,
                                 8, 9, 10, 11, 12, 13, 14, 15);
}

// Async global->LDS copy of 16 bytes (CDNA5, tracked by ASYNCcnt).
// LDS byte address = low 32 bits of the generic pointer (LDS aperture rule).
__device__ __forceinline__ void async_b128(const _Float16* gsrc, _Float16* ldst) {
  unsigned l = (unsigned)(size_t)ldst;
  asm volatile("global_load_async_to_lds_b128 %0, %1, off"
               :: "v"(l), "v"(gsrc) : "memory");
}
__device__ __forceinline__ void wait_async0() {
  asm volatile("s_wait_asynccnt 0x0" ::: "memory");
}

// ---------------------------------------------------------------------------
// Conversion kernels
// ---------------------------------------------------------------------------
__global__ void cvt_f32_to_f16(const float* __restrict__ in,
                               _Float16* __restrict__ out, int n) {
  int i = blockIdx.x * 256 + threadIdx.x;
  if (i < n) out[i] = (_Float16)in[i];
}

// W is [Kd x Nd] row-major (in-features x out-features). Wt[n][k] f16.
__global__ void transpose_cvt(const float* __restrict__ W,
                              _Float16* __restrict__ Wt, int Kd, int Nd) {
  int i = blockIdx.x * 256 + threadIdx.x;
  if (i >= Kd * Nd) return;
  int n = i / Kd, k = i % Kd;
  Wt[(size_t)n * Kd + k] = (_Float16)W[(size_t)k * Nd + n];
}

// ---------------------------------------------------------------------------
// Tiled WMMA GEMM: C[M x N] = A[M x K] * Bt[N x K]^T + bias
// Block tile 128x128, BK=32, 8 waves of 32x64, double-buffered LDS fed by
// async global->LDS DMA. mode 0: f16 head-major [b, h, t, 128] output.
// mode 1: f32 row-major output.
// ---------------------------------------------------------------------------
__global__ __launch_bounds__(256) void gemm_wmma(
    const _Float16* __restrict__ A, const _Float16* __restrict__ Bt,
    const float* __restrict__ bias, void* __restrict__ out,
    int M, int N, int K, int mode, int heads) {
  __shared__ __align__(16) _Float16 sA[2][128 * 40];
  __shared__ __align__(16) _Float16 sB[2][128 * 40];

  const int tid = threadIdx.x, lane = tid & 31, w = tid >> 5;
  const int wm = (w & 3) * 32, wn = (w >> 2) * 64;
  const int bm = blockIdx.y * 128, bn = blockIdx.x * 128;

  // Fixed per-thread staging coordinates (2 x 16B per buffer per matrix).
  const int row0 = tid >> 2, cg0 = (tid & 3) * 8;
  const int row1 = (tid + 256) >> 2, cg1 = ((tid + 256) & 3) * 8;

  auto stage = [&](int buf, int k0) {
    async_b128(&A[(size_t)(bm + row0) * K + k0 + cg0], &sA[buf][row0 * 40 + cg0]);
    async_b128(&Bt[(size_t)(bn + row0) * K + k0 + cg0], &sB[buf][row0 * 40 + cg0]);
    async_b128(&A[(size_t)(bm + row1) * K + k0 + cg1], &sA[buf][row1 * 40 + cg1]);
    async_b128(&Bt[(size_t)(bn + row1) * K + k0 + cg1], &sB[buf][row1 * 40 + cg1]);
  };

  v8f vz;
  for (int i = 0; i < 8; ++i) vz[i] = 0.f;
  v8f acc[2][4];
  for (int i = 0; i < 2; ++i)
    for (int j = 0; j < 4; ++j) acc[i][j] = vz;

  stage(0, 0);
  for (int k0 = 0; k0 < K; k0 += 32) {
    const int buf = (k0 >> 5) & 1;
    wait_async0();          // own stage(i) transfers complete
    __syncthreads();        // everyone's stage(i) visible; buf^1 reads done
    if (k0 + 32 < K) stage(buf ^ 1, k0 + 32);  // DMA next slice during compute

    v16h af[2], bf[4];
    for (int mi = 0; mi < 2; ++mi)
      af[mi] = frag_a(&sA[buf][(wm + mi * 16) * 40], 40, lane);
    for (int ni = 0; ni < 4; ++ni)
      bf[ni] = frag_b(&sB[buf][(wn + ni * 16) * 40], 40, lane);
    for (int mi = 0; mi < 2; ++mi)
      for (int ni = 0; ni < 4; ++ni)
        acc[mi][ni] = wmma_f16(af[mi], bf[ni], acc[mi][ni]);
  }

  const int nl = lane & 15, hi = (lane >> 4) & 1;
  for (int mi = 0; mi < 2; ++mi)
    for (int ni = 0; ni < 4; ++ni) {
      int colb = bn + wn + ni * 16 + nl;
      float bb = bias[colb];
      for (int g = 0; g < 8; ++g) {
        int gm = bm + wm + mi * 16 + g + 8 * hi;   // C layout: M = g + 8*hi
        float val = acc[mi][ni][g] + bb;
        if (mode == 0) {
          int h = colb >> 7, hd = colb & 127;
          int b = gm >> 11, t = gm & 2047;
          ((_Float16*)out)[(((size_t)(b * heads + h)) * 2048 + t) * 128 + hd] =
              (_Float16)val;
        } else {
          ((float*)out)[(size_t)gm * N + colb] = val;
        }
      }
    }
}

// ---------------------------------------------------------------------------
// Attention: per block (b, h, 128 q-rows), 8 waves x 16 q-rows each.
// Online softmax over ALL keys (reference masks AFTER softmax); causality is
// applied by zeroing P for k > q before the PV WMMA. K tile staged by async
// global->LDS DMA; V staged transposed through registers.
// ---------------------------------------------------------------------------
__global__ __launch_bounds__(256) void attn_wmma(
    const _Float16* __restrict__ Q, const _Float16* __restrict__ Kt,
    const _Float16* __restrict__ Vt, _Float16* __restrict__ Z) {
  __shared__ __align__(16) _Float16 sK[64 * 136];   // [key][d]
  __shared__ __align__(16) _Float16 sV[128 * 72];   // transposed: [d][key]
  __shared__ __align__(16) _Float16 sP[8][16 * 72]; // per-wave P tile [q][key]

  const int tid = threadIdx.x, lane = tid & 31, w = tid >> 5;
  const int b = blockIdx.x >> 7;
  const int h = (blockIdx.x >> 4) & 7;
  const int qblk = blockIdx.x & 15;
  const int q0 = qblk * 128 + w * 16;
  const int kvh = h & 3;                            // concat([k,k]): h mod 4

  const _Float16* Qp = Q + (((size_t)(b * 8 + h)) * 2048 + q0) * 128;
  const _Float16* Kp = Kt + ((size_t)(b * 4 + kvh)) * 2048 * 128;
  const _Float16* Vp = Vt + ((size_t)(b * 4 + kvh)) * 2048 * 128;

  const int nl = lane & 15, hi = (lane >> 4) & 1;

  v16h qf[4];                                       // Q 16x128 held in regs
  for (int s = 0; s < 4; ++s) qf[s] = frag_a(Qp + s * 32, 128, lane);

  v8f vz;
  for (int i = 0; i < 8; ++i) vz[i] = 0.f;
  v8f oacc[8];
  for (int i = 0; i < 8; ++i) oacc[i] = vz;
  float mrow[8], lrow[8];
  for (int g = 0; g < 8; ++g) { mrow[g] = -3.0e38f; lrow[g] = 0.f; }

  for (int kb = 0; kb < 2048; kb += 64) {
    __syncthreads();
    // Stage K [64x128] via async DMA; V transposed [128x64] via registers.
    for (int rep = 0; rep < 4; ++rep) {
      int lin = tid + rep * 256;                    // 0..1023
      int key = lin >> 4, d0 = (lin & 15) * 8;
      async_b128(&Kp[((size_t)(kb + key)) * 128 + d0], &sK[key * 136 + d0]);
      v8h vv = *(const v8h*)&Vp[((size_t)(kb + key)) * 128 + d0];
      for (int j = 0; j < 8; ++j) sV[(d0 + j) * 72 + key] = vv[j];
    }
    if (kb + 64 < 2048) {
      __builtin_prefetch(&Kp[((size_t)(kb + 64 + (tid >> 2))) * 128], 0, 1);
      __builtin_prefetch(&Vp[((size_t)(kb + 64 + (tid >> 2))) * 128], 0, 1);
    }
    wait_async0();
    __syncthreads();

    // S = Q K^T: 4 key tiles of 16, reduction over HD=128 in 4 steps of 32.
    v8f sacc[4];
    for (int nt = 0; nt < 4; ++nt) {
      sacc[nt] = vz;
      for (int s = 0; s < 4; ++s) {
        v16h kf = frag_b(&sK[(nt * 16) * 136 + s * 32], 136, lane);
        sacc[nt] = wmma_f16(qf[s], kf, sacc[nt]);
      }
    }

    // Row max over this 64-key strip (rows live across 16-lane groups).
    float bmax[8];
    for (int g = 0; g < 8; ++g) bmax[g] = -3.0e38f;
    for (int nt = 0; nt < 4; ++nt)
      for (int g = 0; g < 8; ++g) bmax[g] = fmaxf(bmax[g], sacc[nt][g]);
    for (int g = 0; g < 8; ++g) {
      float v = bmax[g];
      v = fmaxf(v, __shfl_xor(v, 1));
      v = fmaxf(v, __shfl_xor(v, 2));
      v = fmaxf(v, __shfl_xor(v, 4));
      v = fmaxf(v, __shfl_xor(v, 8));
      bmax[g] = v;
    }
    float scl[8];
    for (int g = 0; g < 8; ++g) {
      float mn = fmaxf(mrow[g], bmax[g]);
      scl[g] = __expf(mrow[g] - mn);
      mrow[g] = mn;
      lrow[g] *= scl[g];
    }
    for (int no = 0; no < 8; ++no)
      for (int g = 0; g < 8; ++g) oacc[no][g] *= scl[g];

    // P = exp(S - m); denominator over ALL keys, PV masked causally.
    float psum[8];
    for (int g = 0; g < 8; ++g) psum[g] = 0.f;
    for (int nt = 0; nt < 4; ++nt) {
      int keyg = kb + nt * 16 + nl;
      for (int g = 0; g < 8; ++g) {
        float pv = __expf(sacc[nt][g] - mrow[g]);
        psum[g] += pv;
        int qg = q0 + g + 8 * hi;
        _Float16 ph = (keyg <= qg) ? (_Float16)pv : (_Float16)0.f;
        sP[w][(g + 8 * hi) * 72 + nt * 16 + nl] = ph;
      }
    }
    for (int g = 0; g < 8; ++g) {
      float v = psum[g];
      v += __shfl_xor(v, 1);
      v += __shfl_xor(v, 2);
      v += __shfl_xor(v, 4);
      v += __shfl_xor(v, 8);
      lrow[g] += v;
    }

    // O += P V : reduction over 64 keys in 2 steps of 32; 8 d-tiles of 16.
    for (int s = 0; s < 2; ++s) {
      v16h pf = frag_a(&sP[w][s * 32], 72, lane);
      for (int no = 0; no < 8; ++no) {
        v16h vf = frag_b(&sV[(no * 16) * 72 + s * 32], 72, lane);
        oacc[no] = wmma_f16(pf, vf, oacc[no]);
      }
    }
  }

  float inv[8];
  for (int g = 0; g < 8; ++g) inv[g] = 1.0f / lrow[g];
  for (int no = 0; no < 8; ++no) {
    int dcol = h * 128 + no * 16 + nl;
    for (int g = 0; g < 8; ++g) {
      int t = q0 + g + 8 * hi;
      Z[(((size_t)b) * 2048 + t) * 1024 + dcol] = (_Float16)(oacc[no][g] * inv[g]);
    }
  }
}

// ---------------------------------------------------------------------------
// Host launcher
// ---------------------------------------------------------------------------
static inline size_t align256(size_t x) { return (x + 255) & ~(size_t)255; }

extern "C" void kernel_launch(void* const* d_in, const int* in_sizes, int n_in,
                              void* d_out, int out_size, void* d_ws, size_t ws_size,
                              hipStream_t stream) {
  (void)in_sizes; (void)n_in; (void)out_size; (void)ws_size;
  const float* x    = (const float*)d_in[0];
  const float* wq_w = (const float*)d_in[1];
  const float* wq_b = (const float*)d_in[2];
  const float* wk_w = (const float*)d_in[3];
  const float* wk_b = (const float*)d_in[4];
  const float* wv_w = (const float*)d_in[5];
  const float* wv_b = (const float*)d_in[6];
  const float* wo_w = (const float*)d_in[7];
  const float* wo_b = (const float*)d_in[8];

  const int M = 4096;  // B*T
  const int D = 1024;

  char* p = (char*)d_ws;
  _Float16* Xh  = (_Float16*)p; p += align256((size_t)M * D * 2);        // x f16
  _Float16* WqT = (_Float16*)p; p += align256((size_t)1024 * 1024 * 2);
  _Float16* WkT = (_Float16*)p; p += align256((size_t)512 * 1024 * 2);
  _Float16* WvT = (_Float16*)p; p += align256((size_t)512 * 1024 * 2);
  _Float16* WoT = (_Float16*)p; p += align256((size_t)1024 * 1024 * 2);
  _Float16* Qh  = (_Float16*)p; p += align256((size_t)2 * 8 * 2048 * 128 * 2);
  _Float16* Kh  = (_Float16*)p; p += align256((size_t)2 * 4 * 2048 * 128 * 2);
  _Float16* Vh  = (_Float16*)p; p += align256((size_t)2 * 4 * 2048 * 128 * 2);
  _Float16* Zh  = (_Float16*)p; p += align256((size_t)M * D * 2);

  // Stage 0: convert inputs to f16 (weights transposed to [n][k]).
  int nx = M * D;
  cvt_f32_to_f16<<<nx / 256, 256, 0, stream>>>(x, Xh, nx);
  transpose_cvt<<<(1024 * 1024) / 256, 256, 0, stream>>>(wq_w, WqT, 1024, 1024);
  transpose_cvt<<<(1024 * 512) / 256, 256, 0, stream>>>(wk_w, WkT, 1024, 512);
  transpose_cvt<<<(1024 * 512) / 256, 256, 0, stream>>>(wv_w, WvT, 1024, 512);
  transpose_cvt<<<(1024 * 1024) / 256, 256, 0, stream>>>(wo_w, WoT, 1024, 1024);

  // Stage 1: QKV projections -> head-major f16.
  gemm_wmma<<<dim3(8, 32), 256, 0, stream>>>(Xh, WqT, wq_b, Qh, M, 1024, 1024, 0, 8);
  gemm_wmma<<<dim3(4, 32), 256, 0, stream>>>(Xh, WkT, wk_b, Kh, M, 512, 1024, 0, 4);
  gemm_wmma<<<dim3(4, 32), 256, 0, stream>>>(Xh, WvT, wv_b, Vh, M, 512, 1024, 0, 4);

  // Stage 2: attention -> Zh [b][t][h*128+d] f16.
  attn_wmma<<<256, 256, 0, stream>>>(Qh, Kh, Vh, Zh);

  // Stage 3: output projection -> f32 out.
  gemm_wmma<<<dim3(8, 32), 256, 0, stream>>>(Zh, WoT, wo_b, d_out, M, 1024, 1024, 1, 0);
}
